// Speech_MSA_v2_41669772706155
// MI455X (gfx1250) — matrix operand order repaired and checked by reading, not compiled, not run on hardware
//
#include <hip/hip_runtime.h>
#include <math.h>

// ---------------------------------------------------------------------------
// Problem constants (from reference)
// ---------------------------------------------------------------------------
#define EMBED_DIM 512
#define NUM_HEADS 8
#define HEAD_DIM  64
#define NUM_WTOK  64
#define LOCAL_SZ  17
#define BATCH     2
#define SEQ_T     (NUM_WTOK + 4096)      // 4160
#define T_FEA     4096
#define CHUNK_N   64                      // number of chunks == NUM_WTOK
#define CHUNK_SZ  64                      // tokens per chunk
#define BH        (BATCH * NUM_HEADS)     // 16
#define M_ROWS    (BATCH * SEQ_T)         // 8320
#define SCALING   0.125f                  // 64^-0.5

typedef __attribute__((ext_vector_type(16))) _Float16 v16h;
typedef __attribute__((ext_vector_type(8)))  _Float16 v8h;
typedef __attribute__((ext_vector_type(4)))  _Float16 v4h;
typedef __attribute__((ext_vector_type(8)))  float    v8f;
typedef __attribute__((ext_vector_type(4)))  float    v4f_t;

// Build a 16-half WMMA fragment from two contiguous 16-byte LDS runs
// (lowers to two ds_load_b128).
__device__ __forceinline__ v16h frag16(const _Float16* p0, const _Float16* p1)
{
    v8h lo = *(const v8h*)p0;
    v8h hi = *(const v8h*)p1;
    return __builtin_shufflevector(lo, hi,
                                   0, 1, 2, 3, 4, 5, 6, 7,
                                   8, 9, 10, 11, 12, 13, 14, 15);
}

__device__ __forceinline__ v4h cvt4(v4f_t v)
{
    v4h h;
    h[0] = (_Float16)v[0]; h[1] = (_Float16)v[1];
    h[2] = (_Float16)v[2]; h[3] = (_Float16)v[3];
    return h;
}

// ===========================================================================
// WMMA GEMM: C[M x NTOT] = A[M x 512] * W[512 x NTOT] + bias
//   MODE 0: scatter into head-major Q (scaled), K, V   (NTOT = 1536)
//   MODE 1: plain row-major store                       (NTOT = 512)
// Block tile 128(M) x 64(N), 256 threads = 8 waves in 4(M) x 2(N) layout.
// Each wave owns a 32x32 slab = four v_wmma_f32_16x16x32_f16 per K-step.
// Software-pipelined, double-buffered LDS: next tile's global loads are
// issued before the current tile's WMMAs; one barrier per K-step.
//  - A tile staged row-major   As[buf][128][32] (frag = 2 contiguous b128)
//  - B tile staged TRANSPOSED  Bst[buf][64][32] via 4x4 register transpose
//    (4 coalesced global_load_b128 along k + 4 packed ds_store_b64)
// ===========================================================================
template <int NTOT, int MODE>
__global__ void __launch_bounds__(256)
gemm_wmma_kernel(const float* __restrict__ A, const float* __restrict__ W,
                 const float* __restrict__ bias,
                 float* __restrict__ O0, float* __restrict__ O1,
                 float* __restrict__ O2)
{
    __shared__ _Float16 As[2][128][32];   // row-major [m][k], 64B row stride
    __shared__ _Float16 Bst[2][64][32];   // TRANSPOSED [n][k], 64B row stride

    const int m0 = blockIdx.x * 128;      // 8320 / 128 = 65 tiles
    const int n0 = blockIdx.y * 64;

    const int tid  = threadIdx.x;
    const int wave = tid >> 5;            // 0..7
    const int lane = tid & 31;
    const int wm   = (wave & 3) * 32;     // wave m-offset in tile
    const int wn   = (wave >> 2) * 32;    // wave n-offset in tile
    const int lr   = lane & 15;
    const int lg   = lane >> 4;           // lane group (0 or 1)

    // B staging assignment (threads 0..127): 4(k) x 4(n) block each
    const int bn4 = (tid & 15) << 2;
    const int bk4 = (tid >> 4) << 2;

    v8f acc[4] = {};                      // [mi*2 + nt]
    v4f_t areg[4];
    v4f_t breg[4];

    // ---- issue global loads for K-step kk into registers ----
    auto load_stage = [&](int kk) {
#pragma unroll
        for (int it = 0; it < 4; ++it) {
            int f  = it * 256 + tid;       // float4 index 0..1023
            int r  = f >> 3;               // 8 float4 per row
            int c4 = (f & 7) << 2;
            areg[it] = *(const v4f_t*)&A[(size_t)(m0 + r) * 512 + kk + c4];
        }
        if (tid < 128) {
#pragma unroll
            for (int j = 0; j < 4; ++j)
                breg[j] = *(const v4f_t*)&W[(size_t)(kk + bk4 + j) * NTOT + n0 + bn4];
        }
    };
    // ---- convert + store staged registers into LDS buffer b ----
    auto store_stage = [&](int b) {
#pragma unroll
        for (int it = 0; it < 4; ++it) {
            int f  = it * 256 + tid;
            int r  = f >> 3;
            int c4 = (f & 7) << 2;
            *(v4h*)&As[b][r][c4] = cvt4(areg[it]);
        }
        if (tid < 128) {
            // register 4x4 transpose -> packed b64 stores
#pragma unroll
            for (int i = 0; i < 4; ++i) {
                v4h h;
                h[0] = (_Float16)breg[0][i]; h[1] = (_Float16)breg[1][i];
                h[2] = (_Float16)breg[2][i]; h[3] = (_Float16)breg[3][i];
                *(v4h*)&Bst[b][bn4 + i][bk4] = h;
            }
        }
    };

    load_stage(0);
    store_stage(0);
    __syncthreads();

    for (int s = 0; s < 16; ++s) {
        const int cur = s & 1;
        // issue next tile's global loads before computing (overlap with WMMA)
        if (s < 15) {
            load_stage((s + 1) * 32);
            if (tid == 0 && s < 14) {
                __builtin_prefetch(&W[(size_t)((s + 2) * 32) * NTOT + n0], 0, 1);
            }
        }

        // ---- fragments (each = two ds_load_b128) ----
        // A lane layout: halves 0..7 -> K = 8g+j ; halves 8..15 -> K = 16+8g+j
        v16h a0 = frag16(&As[cur][wm + lr][8 * lg],
                         &As[cur][wm + lr][16 + 8 * lg]);
        v16h a1 = frag16(&As[cur][wm + 16 + lr][8 * lg],
                         &As[cur][wm + 16 + lr][16 + 8 * lg]);
        // B lane layout: halves 0..7 -> K = 8g+j ; halves 8..15 -> K = 16+8g+j, column = lr
        v16h b0 = frag16(&Bst[cur][wn + lr][8 * lg],
                         &Bst[cur][wn + lr][16 + 8 * lg]);
        v16h b1 = frag16(&Bst[cur][wn + 16 + lr][8 * lg],
                         &Bst[cur][wn + 16 + lr][16 + 8 * lg]);

        acc[0] = __builtin_amdgcn_wmma_f32_16x16x32_f16(
                     false, a0, false, b0, (short)0, acc[0], false, false);
        acc[1] = __builtin_amdgcn_wmma_f32_16x16x32_f16(
                     false, a0, false, b1, (short)0, acc[1], false, false);
        acc[2] = __builtin_amdgcn_wmma_f32_16x16x32_f16(
                     false, a1, false, b0, (short)0, acc[2], false, false);
        acc[3] = __builtin_amdgcn_wmma_f32_16x16x32_f16(
                     false, a1, false, b1, (short)0, acc[3], false, false);

        if (s < 15) store_stage(1 - cur);
        __syncthreads();
    }

    // ---- epilogue: C layout -> M = i + 8*lg, N = lr within each 16x16 ----
#pragma unroll
    for (int mi = 0; mi < 2; ++mi) {
#pragma unroll
        for (int nt = 0; nt < 2; ++nt) {
            v8f c = acc[mi * 2 + nt];
#pragma unroll
            for (int i = 0; i < 8; ++i) {
                int gm = m0 + wm + mi * 16 + i + 8 * lg;
                int gn = n0 + wn + nt * 16 + lr;
                float val = c[i] + bias[gn];
                if (MODE == 0) {
                    // scatter qkv -> head-major [bh][t][d]
                    int which = gn >> 9;          // 0=Q 1=K 2=V
                    int e     = gn & 511;
                    int h     = e >> 6;
                    int d     = e & 63;
                    int bb    = gm / SEQ_T;
                    int t     = gm - bb * SEQ_T;
                    size_t off = (((size_t)(bb * NUM_HEADS + h)) * SEQ_T + t) * 64 + d;
                    if (which == 0)      O0[off] = val * SCALING;
                    else if (which == 1) O1[off] = val;
                    else                 O2[off] = val;
                } else {
                    O0[(size_t)gm * NTOT + gn] = val;
                }
            }
        }
    }
}

// ===========================================================================
// Window-token attention: per (bh, n): online softmax over 65 keys
//   Kcat[c] = (c==0) ? K[bh][n] : K[bh][64 + n*64 + c-1]
// One wave per (bh, n). Lane covers dims {lane, lane+32}.
// Writes out_wtok[bh][n][d] and attn_concat rows t = 0..63.
// ===========================================================================
__global__ void __launch_bounds__(256)
wtok_attn_kernel(const float* __restrict__ Qh, const float* __restrict__ Kh,
                 const float* __restrict__ Vh,
                 float* __restrict__ out_wtok, float* __restrict__ attn)
{
    const int bh   = blockIdx.x;     // 0..15
    const int wave = threadIdx.x >> 5;
    const int lane = threadIdx.x & 31;

    for (int ni = 0; ni < 8; ++ni) {
        const int n = wave * 8 + ni;
        const float* q = Qh + ((size_t)bh * SEQ_T + n) * 64;
        float q0 = q[lane], q1 = q[lane + 32];

        float m = -INFINITY, s = 0.f, acc0 = 0.f, acc1 = 0.f;
        for (int c = 0; c < 65; ++c) {
            int row = (c == 0) ? n : (NUM_WTOK + n * CHUNK_SZ + (c - 1));
            const float* kr = Kh + ((size_t)bh * SEQ_T + row) * 64;
            const float* vr = Vh + ((size_t)bh * SEQ_T + row) * 64;
            float p = q0 * kr[lane] + q1 * kr[lane + 32];
#pragma unroll
            for (int off = 16; off > 0; off >>= 1)
                p += __shfl_xor(p, off, 32);
            float nm    = fmaxf(m, p);
            float scale = __expf(m - nm);      // 0 on first iter (m = -inf)
            float w     = __expf(p - nm);
            s    = s * scale + w;
            acc0 = acc0 * scale + w * vr[lane];
            acc1 = acc1 * scale + w * vr[lane + 32];
            m = nm;
        }
        float o0 = acc0 / s, o1 = acc1 / s;

        size_t woff = ((size_t)bh * NUM_WTOK + n) * 64;
        out_wtok[woff + lane]      = o0;
        out_wtok[woff + lane + 32] = o1;

        int bb = bh >> 3, h = bh & 7;
        size_t coff = ((size_t)bb * SEQ_T + n) * EMBED_DIM + h * 64;
        attn[coff + lane]      = o0;
        attn[coff + lane + 32] = o1;
    }
}

// ===========================================================================
// Local banded attention. One 64-thread block per (chunk, bh).
// Stages 80-row K/V halo + 64-row Q chunk in LDS (float4 vectorized).
// Score slot 0 = Q . out_wtok[chunk]  -> softmax ACROSS the 64-token chunk.
// Slots 1..17 = 17-wide zero-padded local window -> per-token softmax.
// ===========================================================================
#define WEXT 80
__global__ void __launch_bounds__(64)
local_attn_kernel(const float* __restrict__ Qh, const float* __restrict__ Kh,
                  const float* __restrict__ Vh,
                  const float* __restrict__ out_wtok,
                  float* __restrict__ attn)
{
    __shared__ float Ks[WEXT][64];
    __shared__ float Vs[WEXT][64];
    __shared__ float Qs[64][64];
    __shared__ float w0s[64];
    __shared__ float s0buf[64];

    const int chunk = blockIdx.x;    // 0..63
    const int bh    = blockIdx.y;    // 0..15
    const int tid   = threadIdx.x;   // 0..63 (token within chunk)

    // ---- stage K/V halo rows [chunk*64-8, chunk*64+71], zero pad OOB ----
#pragma unroll
    for (int it = 0; it < WEXT * 16 / 64; ++it) {   // float4 granularity
        int i  = it * 64 + tid;
        int r  = i >> 4, d4 = (i & 15) << 2;
        int gt = chunk * CHUNK_SZ - 8 + r;           // feature-token index
        v4f_t kv = {}, vv = {};
        if (gt >= 0 && gt < T_FEA) {
            size_t off = ((size_t)bh * SEQ_T + NUM_WTOK + gt) * 64 + d4;
            kv = *(const v4f_t*)&Kh[off];
            vv = *(const v4f_t*)&Vh[off];
        }
        *(v4f_t*)&Ks[r][d4] = kv;
        *(v4f_t*)&Vs[r][d4] = vv;
    }
    // ---- stage Q chunk ----
#pragma unroll
    for (int it = 0; it < 64 * 16 / 64; ++it) {
        int i = it * 64 + tid;
        int r = i >> 4, d4 = (i & 15) << 2;
        *(v4f_t*)&Qs[r][d4] =
            *(const v4f_t*)&Qh[((size_t)bh * SEQ_T + NUM_WTOK + chunk * CHUNK_SZ + r) * 64 + d4];
    }
    w0s[tid] = out_wtok[((size_t)bh * NUM_WTOK + chunk) * 64 + tid];
    __syncthreads();

    // ---- scores: sc[0] = wtok slot, sc[1..17] = local window ----
    float sc[18];
#pragma unroll
    for (int w = 0; w < 18; ++w) sc[w] = 0.f;
    for (int d = 0; d < 64; ++d) {
        float q = Qs[tid][d];
        sc[0] += q * w0s[d];
#pragma unroll
        for (int w = 0; w < LOCAL_SZ; ++w)
            sc[1 + w] += q * Ks[tid + w][d];     // row t-8+w maps to tid+w
    }
    s0buf[tid] = sc[0];
    __syncthreads();

    // ---- chunk-wide softmax of the wtok score ----
    float m0 = -INFINITY;
    for (int i = 0; i < 64; ++i) m0 = fmaxf(m0, s0buf[i]);
    float sum0 = 0.f;
    for (int i = 0; i < 64; ++i) sum0 += __expf(s0buf[i] - m0);
    float w_wtok = __expf(sc[0] - m0) / sum0;

    // ---- per-token softmax over the 17 local scores ----
    float ml = sc[1];
#pragma unroll
    for (int w = 2; w < 18; ++w) ml = fmaxf(ml, sc[w]);
    float p[LOCAL_SZ], sl = 0.f;
#pragma unroll
    for (int w = 0; w < LOCAL_SZ; ++w) { p[w] = __expf(sc[1 + w] - ml); sl += p[w]; }
    float inv = 1.f / sl;

    // ---- output: w_wtok * w0 + sum_w p[w] * V[t-8+w] ----
    int bb = bh >> 3, h = bh & 7;
    size_t coff = ((size_t)bb * SEQ_T + NUM_WTOK + chunk * CHUNK_SZ + tid) * EMBED_DIM + h * 64;
    for (int d = 0; d < 64; ++d) {
        float o = w_wtok * w0s[d];
#pragma unroll
        for (int w = 0; w < LOCAL_SZ; ++w)
            o += (p[w] * inv) * Vs[tid + w][d];
        attn[coff + d] = o;
    }
}

// ===========================================================================
// Launch
// ===========================================================================
extern "C" void kernel_launch(void* const* d_in, const int* in_sizes, int n_in,
                              void* d_out, int out_size, void* d_ws, size_t ws_size,
                              hipStream_t stream)
{
    const float* x     = (const float*)d_in[0];   // (2, 4160, 512)
    const float* W_qkv = (const float*)d_in[1];   // (512, 1536)
    const float* b_qkv = (const float*)d_in[2];   // (1536,)
    const float* W_out = (const float*)d_in[3];   // (512, 512)
    const float* b_out = (const float*)d_in[4];   // (512,)
    float* out = (float*)d_out;                   // (2, 4160, 512)

    const size_t QKV_ELEMS = (size_t)BH * SEQ_T * 64;   // 4,259,840 each
    float* Qh       = (float*)d_ws;
    float* Kh       = Qh + QKV_ELEMS;
    float* Vh       = Kh + QKV_ELEMS;
    float* out_wtok = Vh + QKV_ELEMS;                   // 16*64*64
    float* attn     = out_wtok + (size_t)BH * NUM_WTOK * 64;  // 2*4160*512

    // 1) QKV projection GEMM (WMMA f16) with head-major scatter epilogue
    {
        dim3 grid(M_ROWS / 128, 1536 / 64);   // 65 x 24
        gemm_wmma_kernel<1536, 0><<<grid, 256, 0, stream>>>(
            x, W_qkv, b_qkv, Qh, Kh, Vh);
    }
    // 2) Window-token attention (fills out_wtok + attn rows t=0..63)
    wtok_attn_kernel<<<BH, 256, 0, stream>>>(Qh, Kh, Vh, out_wtok, attn);

    // 3) Local banded attention (fills attn rows t=64..4159)
    {
        dim3 grid(CHUNK_N, BH);               // 64 x 16
        local_attn_kernel<<<grid, 64, 0, stream>>>(Qh, Kh, Vh, out_wtok, attn);
    }
    // 4) Output projection GEMM (WMMA f16)
    {
        dim3 grid(M_ROWS / 128, 512 / 64);    // 65 x 8
        gemm_wmma_kernel<512, 1><<<grid, 256, 0, stream>>>(
            attn, W_out, b_out, out, nullptr, nullptr);
    }
}
